// GlobalGenerator3_24026047053876
// MI455X (gfx1250) — matrix operand order repaired and checked by reading, not compiled, not run on hardware
//
#include <hip/hip_runtime.h>
#include <hip/hip_bf16.h>

typedef _Float16 f16;
typedef __attribute__((ext_vector_type(16))) _Float16 v16h;
typedef __attribute__((ext_vector_type(8)))  _Float16 v8h;
typedef __attribute__((ext_vector_type(8)))  float    v8f;

#define ACT_NONE  0
#define ACT_RELU  1
#define ACT_LRELU 2

// ---------------------------------------------------------------------------
// f32 -> f16 convert (layout-preserving; used for C==1 tensors)
// ---------------------------------------------------------------------------
__global__ void k_cvt(const float* __restrict__ in, f16* __restrict__ out, int n) {
    int i = blockIdx.x * blockDim.x + threadIdx.x;
    if (i < n) out[i] = (f16)in[i];
}

// NCHW f32 -> NHWC f16
__global__ void k_nchw2nhwc(const float* __restrict__ in, f16* __restrict__ out,
                            int C, int HW) {
    int i = blockIdx.x * blockDim.x + threadIdx.x;
    if (i >= C * HW) return;
    int c = i % C; int p = i / C;
    out[i] = (f16)in[(size_t)c * HW + p];
}

// weights OIHW f32 -> [kh*ks+kw][Co][Ci] f16 (A-rows channel-contiguous)
__global__ void k_wrep(const float* __restrict__ w, f16* __restrict__ out,
                       int Co, int Ci, int ks) {
    int i = blockIdx.x * blockDim.x + threadIdx.x;
    int tot = Co * Ci * ks * ks;
    if (i >= tot) return;
    int ci = i % Ci; int t = i / Ci; int o = t % Co; int kk = t / Co;
    int kh = kk / ks, kw = kk % ks;
    out[i] = (f16)w[(((size_t)o * Ci + ci) * ks + kh) * ks + kw];
}

// NHWC channel concat: out[p][0:Ca]=a, out[p][Ca:Ca+Cb]=b
__global__ void k_cat(const f16* __restrict__ a, const f16* __restrict__ b,
                      f16* __restrict__ out, int Ca, int Cb, int npix) {
    int C = Ca + Cb;
    int i = blockIdx.x * blockDim.x + threadIdx.x;
    if (i >= npix * C) return;
    int c = i % C; int p = i / C;
    out[i] = (c < Ca) ? a[(size_t)p * Ca + c] : b[(size_t)p * Cb + (c - Ca)];
}

// ---------------------------------------------------------------------------
// Implicit-GEMM conv via WMMA f32_16x16x32_f16, NHWC activations.
//   Conv as sum of ks*ks shifted 1x1 GEMMs:  M=Co, N=OH*OW, K=Ci per (kh,kw).
//   wt: [kh*ks+kw][Co][Ci] f16.  in: [H][W][Ci] f16.  out: [OH][OW][Co] f16.
//   One wave computes a 32(M) x 64(N) tile: 2 M-subtiles x 4 N-subtiles
//   (8 f32 accumulators = 64 VGPRs). Each B fragment feeds 2 WMMAs; each
//   c0-step does 8 WMMAs on 6 fragment loads (2 A + 4 B, all 2x b128).
//   Aligned fast path (Ci%32==0): A/B fragments are 2x 16-byte v8h loads.
//   Generic path handles Ci in {1,3}.  refl=1 -> reflect pad (pad==1).
//   Epilogue: +bias, activation, optional NHWC residual, f16 NHWC store and/or
//   f32 NCHW store (final layer).
// ---------------------------------------------------------------------------
__global__ __launch_bounds__(256) void k_conv_wmma(
    const f16* __restrict__ in, const f16* __restrict__ wt,
    const float* __restrict__ bias, const f16* __restrict__ res,
    f16* __restrict__ out, float* __restrict__ out32,
    int Ci, int Co, int H, int W, int OH, int OW,
    int ks, int stride, int pad, int refl, int act)
{
    const int lane = threadIdx.x & 31;
    const int wave = threadIdx.x >> 5;
    const int tiles_m = (Co + 31) >> 5;
    const int npix = OH * OW;
    const int tiles_n = (npix + 63) >> 6;
    const int tile = blockIdx.x * 8 + wave;              // wave-uniform
    if (tile >= tiles_m * tiles_n) return;               // whole-wave exit

    const int tm = tile % tiles_m;
    const int tn = tile / tiles_m;
    const int m0 = tm << 5;
    const int n0 = tn << 6;
    const int col = lane & 15;
    const int half = lane >> 4;

    int  nj[4], oyj[4], oxj[4];
    bool nvj[4];
#pragma unroll
    for (int j = 0; j < 4; ++j) {
        int n = n0 + (j << 4) + col;
        nvj[j] = n < npix;
        nj[j]  = nvj[j] ? n : 0;
        oyj[j] = nj[j] / OW;
        oxj[j] = nj[j] % OW;
    }
    int mrow0 = m0 + col;      if (mrow0 > Co - 1) mrow0 = Co - 1;
    int mrow1 = m0 + 16 + col; if (mrow1 > Co - 1) mrow1 = Co - 1;

    v8f acc[2][4] = {};
    const size_t planeW = (size_t)Co * Ci;               // weight plane per (kh,kw)
    const bool alig = (Ci & 31) == 0;

    if (alig) {
        for (int kh = 0; kh < ks; ++kh)
        for (int kw = 0; kw < ks; ++kw) {
            const f16* __restrict__ wp = wt + (size_t)(kh * ks + kw) * planeW;
            const f16* __restrict__ wk0 = wp + (size_t)mrow0 * Ci;
            const f16* __restrict__ wk1 = wp + (size_t)mrow1 * Ci;
            size_t bb[4]; bool bv[4];
#pragma unroll
            for (int j = 0; j < 4; ++j) {
                int iy = oyj[j] * stride + kh - pad;
                int ix = oxj[j] * stride + kw - pad;
                bool v = nvj[j];
                if (refl) {
                    iy = iy < 0 ? -iy : (iy >= H ? 2 * H - 2 - iy : iy);
                    ix = ix < 0 ? -ix : (ix >= W ? 2 * W - 2 - ix : ix);
                } else {
                    v = v && iy >= 0 && iy < H && ix >= 0 && ix < W;
                }
                bv[j] = v;
                bb[j] = ((size_t)iy * W + ix) * (size_t)Ci;
            }
            for (int c0 = 0; c0 < Ci; c0 += 32) {
                // A fragments: lanes 0-15 carry K c0+0..7 / c0+16..23,
                //              lanes 16-31 carry K c0+8..15 / c0+24..31
                const f16* ap0 = wk0 + c0 + (half << 3);
                v8h a0l = *(const v8h*)ap0;
                v8h a0h = *(const v8h*)(ap0 + 16);
                v16h a0 = __builtin_shufflevector(a0l, a0h,
                    0,1,2,3,4,5,6,7,8,9,10,11,12,13,14,15);
                const f16* ap1 = wk1 + c0 + (half << 3);
                v8h a1l = *(const v8h*)ap1;
                v8h a1h = *(const v8h*)(ap1 + 16);
                v16h a1 = __builtin_shufflevector(a1l, a1h,
                    0,1,2,3,4,5,6,7,8,9,10,11,12,13,14,15);
#pragma unroll
                for (int j = 0; j < 4; ++j) {
                    v8h bl = {}, bh = {};
                    if (bv[j]) {
                        const f16* bp = in + bb[j] + c0 + (half << 3);
                        bl = *(const v8h*)bp;
                        bh = *(const v8h*)(bp + 16);
                    }
                    v16h b = __builtin_shufflevector(bl, bh,
                        0,1,2,3,4,5,6,7,8,9,10,11,12,13,14,15);
                    acc[0][j] = __builtin_amdgcn_wmma_f32_16x16x32_f16(
                        false, a0, false, b, (short)0, acc[0][j], false, false);
                    acc[1][j] = __builtin_amdgcn_wmma_f32_16x16x32_f16(
                        false, a1, false, b, (short)0, acc[1][j], false, false);
                }
            }
        }
    } else {
        // generic scalar-gather path (Ci = 1 or 3 sketch/input convs)
        for (int kh = 0; kh < ks; ++kh)
        for (int kw = 0; kw < ks; ++kw) {
            const f16* __restrict__ wp = wt + (size_t)(kh * ks + kw) * planeW;
            const f16* __restrict__ wk0 = wp + (size_t)mrow0 * Ci;
            const f16* __restrict__ wk1 = wp + (size_t)mrow1 * Ci;
            size_t bb[4]; bool bv[4];
#pragma unroll
            for (int j = 0; j < 4; ++j) {
                int iy = oyj[j] * stride + kh - pad;
                int ix = oxj[j] * stride + kw - pad;
                bool v = nvj[j];
                if (refl) {
                    iy = iy < 0 ? -iy : (iy >= H ? 2 * H - 2 - iy : iy);
                    ix = ix < 0 ? -ix : (ix >= W ? 2 * W - 2 - ix : ix);
                } else {
                    v = v && iy >= 0 && iy < H && ix >= 0 && ix < W;
                }
                bv[j] = v;
                bb[j] = ((size_t)iy * W + ix) * (size_t)Ci;
            }
            for (int c0 = 0; c0 < Ci; c0 += 32) {
                v16h a0, a1;
#pragma unroll
                for (int e = 0; e < 16; ++e) {
                    int ka = c0 + e + (((e >> 3) + half) << 3);
                    a0[e] = (ka < Ci) ? wk0[ka] : (f16)0;
                    a1[e] = (ka < Ci) ? wk1[ka] : (f16)0;
                }
#pragma unroll
                for (int j = 0; j < 4; ++j) {
                    v16h b;
#pragma unroll
                    for (int e = 0; e < 16; ++e) {
                        int kb = c0 + e + (((e >> 3) + half) << 3);
                        b[e] = (bv[j] && kb < Ci) ? in[bb[j] + kb] : (f16)0;
                    }
                    acc[0][j] = __builtin_amdgcn_wmma_f32_16x16x32_f16(
                        false, a0, false, b, (short)0, acc[0][j], false, false);
                    acc[1][j] = __builtin_amdgcn_wmma_f32_16x16x32_f16(
                        false, a1, false, b, (short)0, acc[1][j], false, false);
                }
            }
        }
    }

    // ---- epilogue: C/D row r of subtile mi holds M = m0 + 16*mi + r + 8*half,
    //      N = n0 + 16j + col ----
    const bool fullM = (Co & 15) == 0;
#pragma unroll
    for (int mi = 0; mi < 2; ++mi) {
        const int mbase = m0 + (mi << 4) + (half << 3);
        float bi[8];
#pragma unroll
        for (int r = 0; r < 8; ++r)
            bi[r] = bias[(mbase + r) < Co ? (mbase + r) : (Co - 1)];
#pragma unroll
        for (int j = 0; j < 4; ++j) {
            if (!nvj[j]) continue;
            size_t obase = (size_t)nj[j] * Co + mbase;
            float vr[8];
#pragma unroll
            for (int r = 0; r < 8; ++r) {
                float v = acc[mi][j][r] + bi[r];
                if (act == ACT_RELU)       v = fmaxf(v, 0.f);
                else if (act == ACT_LRELU) v = (v >= 0.f) ? v : 0.1f * v;
                vr[r] = v;
            }
            if (mbase >= Co) continue;   // whole subtile out of range
            if (res) {
                if (fullM && mbase + 8 <= Co) {
                    v8h rv = *(const v8h*)(res + obase);
#pragma unroll
                    for (int r = 0; r < 8; ++r) vr[r] += (float)rv[r];
                } else {
#pragma unroll
                    for (int r = 0; r < 8; ++r)
                        if (mbase + r < Co) vr[r] += (float)res[obase + r];
                }
            }
            if (out) {
                if (fullM && mbase + 8 <= Co) {
                    v8h ov;
#pragma unroll
                    for (int r = 0; r < 8; ++r) ov[r] = (f16)vr[r];
                    *(v8h*)(out + obase) = ov;
                } else {
#pragma unroll
                    for (int r = 0; r < 8; ++r)
                        if (mbase + r < Co) out[obase + r] = (f16)vr[r];
                }
            }
            if (out32) {
#pragma unroll
                for (int r = 0; r < 8; ++r) {
                    int mm = mbase + r;
                    if (mm < Co) out32[(size_t)mm * npix + nj[j]] = vr[r];
                }
            }
        }
    }
}

// ---------------------------------------------------------------------------
// pixel_shuffle (r=2) + ReLU, NHWC: in [H][W][C*4] -> out [2H][2W][C]
// ---------------------------------------------------------------------------
__global__ void k_pixshuf_relu(const f16* __restrict__ in, f16* __restrict__ out,
                               int C, int H, int W) {
    int i = blockIdx.x * blockDim.x + threadIdx.x;
    int OW = W * 2;
    int total = C * 4 * H * W;
    if (i >= total) return;
    int c = i % C; int t = i / C; int ox = t % OW; int oy = t / OW;
    int ic = c * 4 + (oy & 1) * 2 + (ox & 1);
    float v = (float)in[((size_t)(oy >> 1) * W + (ox >> 1)) * (C * 4) + ic];
    out[i] = (f16)fmaxf(v, 0.f);
}

// nearest resize, NHWC (torch 'nearest': idx = floor(out_i * in / out))
__global__ void k_nearest(const f16* __restrict__ in, f16* __restrict__ out,
                          int C, int IH, int IW, int OH, int OW) {
    int i = blockIdx.x * blockDim.x + threadIdx.x;
    if (i >= C * OH * OW) return;
    int c = i % C; int t = i / C; int ox = t % OW; int oy = t / OW;
    int iy = (oy * IH) / OH;
    int ix = (ox * IW) / OW;
    out[i] = in[((size_t)iy * IW + ix) * C + c];
}

// kernel2d_conv (FAC) NHWC: feat [H][W][C], kern [H][W][C*9] (c*9+kh*3+kw)
__global__ void k_fac(const f16* __restrict__ feat, const f16* __restrict__ kern,
                      f16* __restrict__ out, int C, int H, int W) {
    int i = blockIdx.x * blockDim.x + threadIdx.x;
    if (i >= C * H * W) return;
    int c = i % C; int p = i / C; int x = p % W; int y = p / W;
    float acc = 0.f;
#pragma unroll
    for (int kh = 0; kh < 3; ++kh) {
#pragma unroll
        for (int kw = 0; kw < 3; ++kw) {
            int iy = y + kh - 1; iy = iy < 0 ? 0 : (iy >= H ? H - 1 : iy);
            int ix = x + kw - 1; ix = ix < 0 ? 0 : (ix >= W ? W - 1 : ix);
            acc += (float)feat[((size_t)iy * W + ix) * C + c] *
                   (float)kern[(size_t)p * (C * 9) + c * 9 + kh * 3 + kw];
        }
    }
    out[i] = (f16)acc;
}

// InstanceNorm stats, NHWC: one block per channel
__global__ void k_istats(const f16* __restrict__ x, float* __restrict__ mean,
                         float* __restrict__ rstd, int C, int HW) {
    int c = blockIdx.x;
    float s = 0.f, s2 = 0.f;
    for (int i = threadIdx.x; i < HW; i += blockDim.x) {
        float v = (float)x[(size_t)i * C + c]; s += v; s2 += v * v;
    }
    __shared__ float sh[256], sh2[256];
    sh[threadIdx.x] = s; sh2[threadIdx.x] = s2;
    __syncthreads();
    for (int k = blockDim.x >> 1; k > 0; k >>= 1) {
        if (threadIdx.x < k) { sh[threadIdx.x] += sh[threadIdx.x + k]; sh2[threadIdx.x] += sh2[threadIdx.x + k]; }
        __syncthreads();
    }
    if (threadIdx.x == 0) {
        float mu = sh[0] / (float)HW;
        float var = sh2[0] / (float)HW - mu * mu;
        mean[c] = mu;
        rstd[c] = rsqrtf(var + 1e-5f);
    }
}

// SPADE apply, NHWC: out = (x - mu)*rstd * (1 + g) + b  (in-place safe)
__global__ void k_spade(const f16* __restrict__ x, const float* __restrict__ mean,
                        const float* __restrict__ rstd, const f16* __restrict__ g,
                        const f16* __restrict__ b, f16* __restrict__ out,
                        int C, int HW) {
    int i = blockIdx.x * blockDim.x + threadIdx.x;
    if (i >= C * HW) return;
    int c = i % C;
    float xn = ((float)x[i] - mean[c]) * rstd[c];
    out[i] = (f16)(xn * (1.f + (float)g[i]) + (float)b[i]);
}

// ---------------------------------------------------------------------------
// Host side
// ---------------------------------------------------------------------------
// d_in flattening (setup_inputs insertion order): input, sketch, then each
// param entry contributes w then b.
enum {
    IDX_M1 = 2, IDX_M2 = 4, IDX_M3 = 6, IDX_R0 = 8 /* r{i}a at 8+4i, r{i}b at 10+4i */,
    IDX_DC1 = 44, IDX_DC11 = 46, IDX_C11 = 48,
    IDX_F1C0 = 50, IDX_F1R0 = 52, IDX_F1R1 = 54, IDX_F1C2 = 56,
    IDX_N1SH = 58, IDX_N1G = 60, IDX_N1B = 62,
    IDX_DC2 = 64, IDX_DC22 = 66, IDX_C21 = 68,
    IDX_F2C0 = 70, IDX_F2R0 = 72, IDX_F2R1 = 74, IDX_F2C2 = 76,
    IDX_N2SH = 78, IDX_N2G = 80, IDX_N2B = 82,
    IDX_T0 = 84, IDX_T1 = 86
};

extern "C" void kernel_launch(void* const* d_in, const int* in_sizes, int n_in,
                              void* d_out, int out_size, void* d_ws, size_t ws_size,
                              hipStream_t stream) {
    (void)n_in; (void)out_size; (void)ws_size;
    const int Hf = 256, HWf = 256 * 256;
    const int H2 = 128, HW2 = 128 * 128;
    const int H4 = 64,  HW4 = 64 * 64;

    char* wsp = (char*)d_ws;
    size_t off = 0;
    auto a16 = [&](size_t nelem) -> f16* {
        f16* p = (f16*)(wsp + off);
        off += ((nelem * sizeof(f16) + 255) / 256) * 256;
        return p;
    };
    auto a32 = [&](size_t nelem) -> float* {
        float* p = (float*)(wsp + off);
        off += ((nelem * sizeof(float) + 255) / 256) * 256;
        return p;
    };
    auto ew = [&](long long total) { return dim3((unsigned)((total + 255) / 256)); };

    // generic conv launcher (repacks weights to [kh*ks+kw][Co][Ci] f16 first)
    auto conv = [&](const f16* in, int widx, const f16* res, f16* out, float* out32,
                    int Ci, int Co, int H, int W, int ks, int stride, int pad,
                    int refl, int act) {
        int wn = in_sizes[widx];
        f16* w = a16((size_t)wn);
        k_wrep<<<ew(wn), 256, 0, stream>>>((const float*)d_in[widx], w, Co, Ci, ks);
        const float* b = (const float*)d_in[widx + 1];
        int OH = (H + 2 * pad - ks) / stride + 1;
        int OW = (W + 2 * pad - ks) / stride + 1;
        long long tiles = (long long)((Co + 31) / 32) * ((OH * OW + 63) / 64);
        int blocks = (int)((tiles + 7) / 8);
        k_conv_wmma<<<blocks, 256, 0, stream>>>(in, w, b, res, out, out32,
                                                Ci, Co, H, W, OH, OW,
                                                ks, stride, pad, refl, act);
    };

    // ---- inputs -> NHWC f16 ----
    f16* in16 = a16((size_t)3 * HWf);
    k_nchw2nhwc<<<ew(3 * HWf), 256, 0, stream>>>((const float*)d_in[0], in16, 3, HWf);
    f16* sk16 = a16((size_t)HWf);
    k_cvt<<<ew(HWf), 256, 0, stream>>>((const float*)d_in[1], sk16, HWf);

    // ---- encoder ----
    f16* f1 = a16((size_t)64 * HWf);
    conv(in16, IDX_M1, nullptr, f1, nullptr, 3, 64, Hf, Hf, 3, 1, 1, 0, ACT_RELU);
    f16* f2 = a16((size_t)128 * HW2);
    conv(f1, IDX_M2, nullptr, f2, nullptr, 64, 128, Hf, Hf, 3, 2, 1, 0, ACT_RELU);
    f16* f3 = a16((size_t)256 * HW4);
    conv(f2, IDX_M3, nullptr, f3, nullptr, 128, 256, H2, H2, 3, 2, 1, 0, ACT_RELU);

    // ---- 9 resnet blocks (reflect pad) ----
    f16* xa = a16((size_t)256 * HW4);
    f16* xb = a16((size_t)256 * HW4);
    f16* yt = a16((size_t)256 * HW4);
    const f16* xc = f3;
    for (int i = 0; i < 9; ++i) {
        int wi = IDX_R0 + 4 * i;
        conv(xc, wi, nullptr, yt, nullptr, 256, 256, H4, H4, 3, 1, 1, 1, ACT_RELU);
        f16* xn = (xc == xa) ? xb : xa;
        conv(yt, wi + 2, xc, xn, nullptr, 256, 256, H4, H4, 3, 1, 1, 1, ACT_NONE);
        xc = xn;
    }

    // ---- f4 = concat(x, f3); deconv1 + pixel_shuffle + relu ----
    f16* f4 = a16((size_t)512 * HW4);
    k_cat<<<ew((long long)512 * HW4), 256, 0, stream>>>(xc, f3, f4, 256, 256, HW4);
    f16* d1 = a16((size_t)512 * HW4);
    conv(f4, IDX_DC1, nullptr, d1, nullptr, 512, 512, H4, H4, 3, 1, 1, 0, ACT_NONE);
    f16* feat1 = a16((size_t)128 * HW2);
    k_pixshuf_relu<<<ew((long long)128 * HW2), 256, 0, stream>>>(d1, feat1, 128, H4, H4);

    // ---- stage-1 FAC branch ----
    f16* sk2 = a16((size_t)HW2);
    k_nearest<<<ew(HW2), 256, 0, stream>>>(sk16, sk2, 1, Hf, Hf, H2, H2);
    f16* skf1 = a16((size_t)128 * HW2);
    conv(sk2, IDX_C11, nullptr, skf1, nullptr, 1, 128, H2, H2, 3, 1, 1, 0, ACT_LRELU);
    f16* y1 = a16((size_t)128 * HW2);
    conv(skf1, IDX_F1C0, nullptr, y1, nullptr, 128, 128, H2, H2, 3, 1, 1, 0, ACT_LRELU);
    f16* t1 = a16((size_t)128 * HW2);
    conv(y1, IDX_F1R0, nullptr, t1, nullptr, 128, 128, H2, H2, 3, 1, 1, 0, ACT_LRELU);
    f16* z1 = a16((size_t)128 * HW2);
    conv(t1, IDX_F1R1, y1, z1, nullptr, 128, 128, H2, H2, 3, 1, 1, 0, ACT_NONE);
    f16* kern1 = a16((size_t)1152 * HW2);
    conv(z1, IDX_F1C2, nullptr, kern1, nullptr, 128, 1152, H2, H2, 1, 1, 0, 0, ACT_LRELU);
    f16* fe1 = a16((size_t)128 * HW2);
    k_fac<<<ew((long long)128 * HW2), 256, 0, stream>>>(feat1, kern1, fe1, 128, H2, H2);

    // ---- SPADE norm1 ----
    float* mu1 = a32(128); float* rs1 = a32(128);
    k_istats<<<128, 256, 0, stream>>>(fe1, mu1, rs1, 128, HW2);
    f16* sa1 = a16((size_t)128 * HW2);
    conv(sk2, IDX_N1SH, nullptr, sa1, nullptr, 1, 128, H2, H2, 3, 1, 1, 0, ACT_RELU);
    f16* g1 = a16((size_t)128 * HW2);
    conv(sa1, IDX_N1G, nullptr, g1, nullptr, 128, 128, H2, H2, 3, 1, 1, 0, ACT_NONE);
    f16* b1 = a16((size_t)128 * HW2);
    conv(sa1, IDX_N1B, nullptr, b1, nullptr, 128, 128, H2, H2, 3, 1, 1, 0, ACT_NONE);
    k_spade<<<ew((long long)128 * HW2), 256, 0, stream>>>(fe1, mu1, rs1, g1, b1, fe1, 128, HW2);
    // feat = feat + relu(conv(fe, deconv11))
    f16* feat1b = a16((size_t)128 * HW2);
    conv(fe1, IDX_DC11, feat1, feat1b, nullptr, 128, 128, H2, H2, 3, 1, 1, 0, ACT_RELU);

    // ---- stage 2: concat + deconv2 + pixel_shuffle ----
    f16* cat1 = a16((size_t)256 * HW2);
    k_cat<<<ew((long long)256 * HW2), 256, 0, stream>>>(feat1b, f2, cat1, 128, 128, HW2);
    f16* d2 = a16((size_t)256 * HW2);
    conv(cat1, IDX_DC2, nullptr, d2, nullptr, 256, 256, H2, H2, 3, 1, 1, 0, ACT_NONE);
    f16* feat2 = a16((size_t)64 * HWf);
    k_pixshuf_relu<<<ew((long long)64 * HWf), 256, 0, stream>>>(d2, feat2, 64, H2, H2);

    // ---- stage-2 FAC branch (sketch resize to 256x256 is identity) ----
    f16* skf2 = a16((size_t)64 * HWf);
    conv(sk16, IDX_C21, nullptr, skf2, nullptr, 1, 64, Hf, Hf, 3, 1, 1, 0, ACT_LRELU);
    f16* y2 = a16((size_t)64 * HWf);
    conv(skf2, IDX_F2C0, nullptr, y2, nullptr, 64, 64, Hf, Hf, 3, 1, 1, 0, ACT_LRELU);
    f16* t2 = a16((size_t)64 * HWf);
    conv(y2, IDX_F2R0, nullptr, t2, nullptr, 64, 64, Hf, Hf, 3, 1, 1, 0, ACT_LRELU);
    f16* z2 = a16((size_t)64 * HWf);
    conv(t2, IDX_F2R1, y2, z2, nullptr, 64, 64, Hf, Hf, 3, 1, 1, 0, ACT_NONE);
    f16* kern2 = a16((size_t)576 * HWf);
    conv(z2, IDX_F2C2, nullptr, kern2, nullptr, 64, 576, Hf, Hf, 1, 1, 0, 0, ACT_LRELU);
    f16* fe2 = a16((size_t)64 * HWf);
    k_fac<<<ew((long long)64 * HWf), 256, 0, stream>>>(feat2, kern2, fe2, 64, Hf, Hf);

    // ---- SPADE norm2 ----
    float* mu2 = a32(64); float* rs2 = a32(64);
    k_istats<<<64, 256, 0, stream>>>(fe2, mu2, rs2, 64, HWf);
    f16* sa2 = a16((size_t)128 * HWf);
    conv(sk16, IDX_N2SH, nullptr, sa2, nullptr, 1, 128, Hf, Hf, 3, 1, 1, 0, ACT_RELU);
    f16* g2 = a16((size_t)64 * HWf);
    conv(sa2, IDX_N2G, nullptr, g2, nullptr, 128, 64, Hf, Hf, 3, 1, 1, 0, ACT_NONE);
    f16* b2 = a16((size_t)64 * HWf);
    conv(sa2, IDX_N2B, nullptr, b2, nullptr, 128, 64, Hf, Hf, 3, 1, 1, 0, ACT_NONE);
    k_spade<<<ew((long long)64 * HWf), 256, 0, stream>>>(fe2, mu2, rs2, g2, b2, fe2, 64, HWf);
    f16* feat2b = a16((size_t)64 * HWf);
    conv(fe2, IDX_DC22, feat2, feat2b, nullptr, 64, 64, Hf, Hf, 3, 1, 1, 0, ACT_RELU);

    // ---- tail ----
    f16* cat2 = a16((size_t)128 * HWf);
    k_cat<<<ew((long long)128 * HWf), 256, 0, stream>>>(feat2b, f1, cat2, 64, 64, HWf);
    f16* t0 = a16((size_t)64 * HWf);
    conv(cat2, IDX_T0, nullptr, t0, nullptr, 128, 64, Hf, Hf, 3, 1, 1, 0, ACT_RELU);
    // final 1x1 conv writes f32 NCHW directly into d_out
    conv(t0, IDX_T1, nullptr, nullptr, (float*)d_out, 64, 3, Hf, Hf, 1, 1, 0, 0, ACT_NONE);
}
